// TtFalconAttention_11450382811573
// MI455X (gfx1250) — matrix operand order repaired and checked by reading, not compiled, not run on hardware
//
#include <hip/hip_runtime.h>
#include <hip/hip_bf16.h>

// ---------------------------------------------------------------------------
// Falcon attention block on gfx1250 (MI455X), f16 WMMA path, f32 accumulate.
// GEMMs: double-buffered LDS fed by global_load_async_to_lds_b128 (ASYNCcnt).
// ---------------------------------------------------------------------------

typedef _Float16 f16;
typedef _Float16 h4  __attribute__((ext_vector_type(4)));
typedef _Float16 h8  __attribute__((ext_vector_type(8)));
typedef _Float16 h16 __attribute__((ext_vector_type(16)));
typedef float    f8  __attribute__((ext_vector_type(8)));

#define HIDDEN 4544
#define NHEADS 71
#define HD     64
#define SEQ    1024
#define QKVF   ((NHEADS + 2) * HD)   // 4672

static __device__ __forceinline__ h16 cat8(h8 lo, h8 hi) {
    return __builtin_shufflevector(lo, hi, 0,1,2,3,4,5,6,7,8,9,10,11,12,13,14,15);
}

static __device__ __forceinline__ f8 wmma_f16(h16 a, h16 b, f8 c) {
    // D = A(16x32 f16) * B(32x16 f16) + C(16x16 f32)
    return __builtin_amdgcn_wmma_f32_16x16x32_f16(false, a, false, b,
                                                  (short)0, c, false, false);
}

// Async copy 16B: global (64-bit vaddr) -> LDS (32-bit dsaddr), ASYNCcnt path.
static __device__ __forceinline__ void async_copy_b128(unsigned lds_off,
                                                       const f16* gptr) {
    asm volatile("global_load_async_to_lds_b128 %0, %1, off"
                 :
                 : "v"(lds_off), "v"((unsigned long long)(size_t)gptr)
                 : "memory");
}
static __device__ __forceinline__ void wait_asynccnt0() {
    asm volatile("s_wait_asynccnt 0x0" ::: "memory");
}

// ---------------------------------------------------------------------------
// f32 -> f16 elementwise (vectorized), n4 = count/4
// ---------------------------------------------------------------------------
__global__ void cvt_f32_to_f16(const float* __restrict__ in,
                               f16* __restrict__ out, int n4) {
    int i = blockIdx.x * blockDim.x + threadIdx.x;
    if (i >= n4) return;
    float4 v = ((const float4*)in)[i];
    h4 r;
    r[0] = (f16)v.x; r[1] = (f16)v.y; r[2] = (f16)v.z; r[3] = (f16)v.w;
    ((h4*)out)[i] = r;
}

// ---------------------------------------------------------------------------
// Transpose + convert: in f32 [K][N] -> out f16 [N][K]  (32x32 LDS tiles)
// ---------------------------------------------------------------------------
__global__ void transpose_cvt(const float* __restrict__ in,
                              f16* __restrict__ out, int K, int N) {
    __shared__ float tile[32][33];
    const int tx = threadIdx.x & 31;
    const int ty = threadIdx.x >> 5;              // 0..7
    const int kb = blockIdx.y * 32, nb = blockIdx.x * 32;
    for (int r = ty; r < 32; r += 8)
        tile[r][tx] = in[(size_t)(kb + r) * N + nb + tx];
    __syncthreads();
    for (int r = ty; r < 32; r += 8)
        out[(size_t)(nb + r) * K + kb + tx] = (f16)tile[tx][r];
}

// ---------------------------------------------------------------------------
// GEMM: C[M][N] f32 = A[M][K] f16 (row-major) * Bt[N][K] f16 (B col-major)
// Block tile 256x64, BK=32, 256 threads (8 waves); wave w -> rows w*32..+32
// (two 16-row WMMA slabs) x 64 cols (4 WMMA n-tiles) => 8 WMMA / K-step.
// Double-buffered LDS, filled with async global->LDS b128 copies; one
// s_wait_asynccnt + one barrier per K-step. LDS rows padded to 48 halves.
// ---------------------------------------------------------------------------
__global__ __launch_bounds__(256) void gemm_wmma(
    const f16* __restrict__ A, const f16* __restrict__ Bt,
    float* __restrict__ C, int M, int N, int K) {
    __shared__ __align__(16) f16 As[2][256 * 48];
    __shared__ __align__(16) f16 Bs[2][64 * 48];
    const int tid  = threadIdx.x;
    const int lane = tid & 31;
    const int w    = tid >> 5;
    const int lm = lane & 15, lh = lane >> 4;
    const int rowBase = blockIdx.y * 256;
    const int colBase = blockIdx.x * 64;
    // loaders: A -> one 32-half row per thread (4x b128); B -> 8 halves/thread
    const int br = tid >> 2, bo = (tid & 3) * 8;
    const f16* agp = A  + (size_t)(rowBase + tid) * K;
    const f16* bgp = Bt + (size_t)(colBase + br) * K + bo;
    const unsigned aLds[2] = { (unsigned)(size_t)&As[0][tid * 48],
                               (unsigned)(size_t)&As[1][tid * 48] };
    const unsigned bLds[2] = { (unsigned)(size_t)&Bs[0][br * 48 + bo],
                               (unsigned)(size_t)&Bs[1][br * 48 + bo] };

    f8 acc[2][4] = {};

    // prologue: stage tile 0
    {
        const f16* ag = agp;
        #pragma unroll
        for (int c = 0; c < 4; ++c) async_copy_b128(aLds[0] + c * 16, ag + c * 8);
        async_copy_b128(bLds[0], bgp);
        wait_asynccnt0();
        __syncthreads();
    }

    const int nsteps = K >> 5;                // K / 32
    for (int s = 0; s < nsteps; ++s) {
        const int p = s & 1;
        // stage next tile into the other buffer (overlaps with WMMA below)
        if (s + 1 < nsteps) {
            const int k1 = (s + 1) << 5;
            const f16* ag = agp + k1;
            #pragma unroll
            for (int c = 0; c < 4; ++c)
                async_copy_b128(aLds[1 - p] + c * 16, ag + c * 8);
            async_copy_b128(bLds[1 - p], bgp + k1);
        }
        // compute on current buffer
        h16 af[2];
        #pragma unroll
        for (int r = 0; r < 2; ++r) {
            const f16* ap = &As[p][(w * 32 + r * 16 + lm) * 48 + lh * 8];
            af[r] = cat8(*(const h8*)ap, *(const h8*)(ap + 16));
        }
        #pragma unroll
        for (int t = 0; t < 4; ++t) {
            const f16* bp = &Bs[p][(t * 16 + lm) * 48 + lh * 8];
            h16 bf = cat8(*(const h8*)bp, *(const h8*)(bp + 16));
            #pragma unroll
            for (int r = 0; r < 2; ++r)
                acc[r][t] = wmma_f16(af[r], bf, acc[r][t]);
        }
        wait_asynccnt0();
        __syncthreads();
    }

    #pragma unroll
    for (int r = 0; r < 2; ++r)
        #pragma unroll
        for (int t = 0; t < 4; ++t)
            #pragma unroll
            for (int v = 0; v < 8; ++v) {
                int row = rowBase + w * 32 + r * 16 + v + 8 * lh;
                int col = colBase + t * 16 + lm;
                C[(size_t)row * N + col] = acc[r][t][v];
            }
}

// ---------------------------------------------------------------------------
// RoPE + QKV split. fused f32 [S][4672] ->
//   qh f16 [71][S][64] (rope, *0.125), kh f16 [S][64] (rope), vt f16 [64][S]
// ---------------------------------------------------------------------------
__global__ void rope_split(const float* __restrict__ fused,
                           f16* __restrict__ qh, f16* __restrict__ kh,
                           f16* __restrict__ vt) {
    const int s = blockIdx.x;
    const int tid = threadIdx.x;
    const int d = tid & 63;
    const float invf = __expf(-(float)(d & 31) * (9.210340371976184f / 32.0f));
    const float ang = (float)s * invf;
    float si, c;
    __sincosf(ang, &si, &c);
    const float* row = fused + (size_t)s * QKVF;
    for (int h = tid >> 6; h < NHEADS; h += 4) {
        const float* qp = row + h * HD;
        float x = qp[d];
        float r = (d < 32) ? -qp[d + 32] : qp[d - 32];
        qh[((size_t)h * SEQ + s) * HD + d] = (f16)((x * c + r * si) * 0.125f);
    }
    if (tid < 64) {
        const float* kp = row + NHEADS * HD;
        float x = kp[d];
        float r = (d < 32) ? -kp[d + 32] : kp[d - 32];
        kh[(size_t)s * HD + d] = (f16)(x * c + r * si);
        vt[(size_t)d * SEQ + s] = (f16)row[(NHEADS + 1) * HD + d];
    }
}

// ---------------------------------------------------------------------------
// Flash attention, one wave per (head, 16-query block), causal, online softmax.
// scores = (Q*0.125) K^T via WMMA; P re-laid out through LDS; ctx = P V.
// ---------------------------------------------------------------------------
__global__ __launch_bounds__(32) void flash_attn(
    const f16* __restrict__ qh, const f16* __restrict__ kh,
    const f16* __restrict__ vt, f16* __restrict__ ctx) {
    const int head = blockIdx.x >> 6;          // / (SEQ/16)
    const int qb   = (blockIdx.x & 63) * 16;
    const int lane = threadIdx.x;
    const int lm = lane & 15, lh = lane >> 4;

    // Q A-fragments for d-halves 0/1, live for the whole kernel
    h16 qa[2];
    {
        const f16* qp = qh + ((size_t)head * SEQ + qb + lm) * HD;
        #pragma unroll
        for (int dh = 0; dh < 2; ++dh) {
            const f16* p = qp + dh * 32 + lh * 8;
            qa[dh] = cat8(*(const h8*)p, *(const h8*)(p + 16));
        }
    }

    f8 O[4] = {};
    float mrow[8], lrow[8];
    #pragma unroll
    for (int v = 0; v < 8; ++v) { mrow[v] = -1e30f; lrow[v] = 0.0f; }

    __shared__ __align__(16) f16 pbuf[16 * 48];

    const int nk = qb + 16;                     // causal key limit (exclusive)
    for (int kb = 0; kb < nk; kb += 32) {
        // ---- scores: 16 q-rows x 32 keys (two 16x16 C tiles) ----
        f8 sc[2] = {};
        #pragma unroll
        for (int t = 0; t < 2; ++t) {
            const f16* kp = kh + (size_t)(kb + t * 16 + lm) * HD;
            #pragma unroll
            for (int dh = 0; dh < 2; ++dh) {
                const f16* p = kp + dh * 32 + lh * 8;
                h16 bf = cat8(*(const h8*)p, *(const h8*)(p + 16));
                sc[t] = wmma_f16(qa[dh], bf, sc[t]);
            }
        }
        // ---- causal mask + per-row max of the new chunk ----
        float mnew[8];
        #pragma unroll
        for (int v = 0; v < 8; ++v) {
            const int row = qb + v + 8 * lh;
            if (kb + lm > row)      sc[0][v] = -1e30f;
            if (kb + 16 + lm > row) sc[1][v] = -1e30f;
            mnew[v] = fmaxf(sc[0][v], sc[1][v]);
        }
        #pragma unroll
        for (int off = 1; off < 16; off <<= 1)
            #pragma unroll
            for (int v = 0; v < 8; ++v)
                mnew[v] = fmaxf(mnew[v], __shfl_xor(mnew[v], off, 16));
        // ---- online softmax update ----
        float rsum[8];
        #pragma unroll
        for (int v = 0; v < 8; ++v) {
            float mn = fmaxf(mrow[v], mnew[v]);
            float alpha = __expf(mrow[v] - mn);
            mrow[v] = mn;
            #pragma unroll
            for (int t = 0; t < 4; ++t) O[t][v] *= alpha;
            float p0 = __expf(sc[0][v] - mn);
            float p1 = __expf(sc[1][v] - mn);
            sc[0][v] = p0; sc[1][v] = p1;
            rsum[v] = p0 + p1;
            lrow[v] *= alpha;
        }
        #pragma unroll
        for (int off = 1; off < 16; off <<= 1)
            #pragma unroll
            for (int v = 0; v < 8; ++v)
                rsum[v] += __shfl_xor(rsum[v], off, 16);
        #pragma unroll
        for (int v = 0; v < 8; ++v) lrow[v] += rsum[v];

        // ---- P (C layout) -> LDS -> A-fragment ----
        __syncthreads();
        #pragma unroll
        for (int t = 0; t < 2; ++t)
            #pragma unroll
            for (int v = 0; v < 8; ++v)
                pbuf[(v + 8 * lh) * 48 + t * 16 + lm] = (f16)sc[t][v];
        __syncthreads();
        h16 pa;
        {
            const f16* p = &pbuf[lm * 48 + lh * 8];
            pa = cat8(*(const h8*)p, *(const h8*)(p + 16));
        }
        // ---- ctx += P * V  (V^T stored [64][SEQ]: table-order-K B frags) ----
        #pragma unroll
        for (int t = 0; t < 4; ++t) {
            const f16* vp = vt + (size_t)(t * 16 + lm) * SEQ + kb + lh * 8;
            h16 bf = cat8(*(const h8*)vp, *(const h8*)(vp + 16));
            O[t] = wmma_f16(pa, bf, O[t]);
        }
    }
    // ---- normalize and store ctx f16 [S][HIDDEN] ----
    #pragma unroll
    for (int v = 0; v < 8; ++v) {
        float inv = 1.0f / lrow[v];
        const int row = qb + v + 8 * lh;
        #pragma unroll
        for (int t = 0; t < 4; ++t)
            ctx[(size_t)row * HIDDEN + head * HD + t * 16 + lm] =
                (f16)(O[t][v] * inv);
    }
}

// ---------------------------------------------------------------------------
extern "C" void kernel_launch(void* const* d_in, const int* in_sizes, int n_in,
                              void* d_out, int out_size, void* d_ws, size_t ws_size,
                              hipStream_t stream) {
    (void)in_sizes; (void)n_in; (void)out_size; (void)ws_size;
    const float* x    = (const float*)d_in[0];   // [1,1,1024,4544]
    // d_in[1] = attention_mask: causal triu(-1e9), applied analytically
    const float* qkvw = (const float*)d_in[2];   // [4544,4672]
    const float* dw   = (const float*)d_in[3];   // [4544,4544]
    float* out = (float*)d_out;                  // [1,1,1024,4544]

    char* ws = (char*)d_ws;
    auto alloc = [&](size_t bytes) -> char* {
        char* p = ws;
        ws += (bytes + 255) & ~(size_t)255;
        return p;
    };
    f16*   xh    = (f16*)  alloc((size_t)SEQ * HIDDEN * 2);
    f16*   qkvT  = (f16*)  alloc((size_t)QKVF * HIDDEN * 2);
    f16*   denT  = (f16*)  alloc((size_t)HIDDEN * HIDDEN * 2);
    float* fused = (float*)alloc((size_t)SEQ * QKVF * 4);
    f16*   qh    = (f16*)  alloc((size_t)NHEADS * SEQ * HD * 2);
    f16*   kh    = (f16*)  alloc((size_t)SEQ * HD * 2);
    f16*   vt    = (f16*)  alloc((size_t)HD * SEQ * 2);
    f16*   ctxh  = (f16*)  alloc((size_t)SEQ * HIDDEN * 2);

    {   // activations f32 -> f16
        int n4 = SEQ * HIDDEN / 4;
        cvt_f32_to_f16<<<n4 / 256, 256, 0, stream>>>(x, xh, n4);
    }
    // weights f32 [K][N] -> f16 [N][K]
    transpose_cvt<<<dim3(QKVF / 32, HIDDEN / 32), 256, 0, stream>>>(qkvw, qkvT, HIDDEN, QKVF);
    transpose_cvt<<<dim3(HIDDEN / 32, HIDDEN / 32), 256, 0, stream>>>(dw, denT, HIDDEN, HIDDEN);
    // QKV projection
    gemm_wmma<<<dim3(QKVF / 64, SEQ / 256), 256, 0, stream>>>(xh, qkvT, fused, SEQ, QKVF, HIDDEN);
    // RoPE + split
    rope_split<<<SEQ, 256, 0, stream>>>(fused, qh, kh, vt);
    // attention
    flash_attn<<<NHEADS * (SEQ / 16), 32, 0, stream>>>(qh, kh, vt, ctxh);
    // output projection
    gemm_wmma<<<dim3(HIDDEN / 64, SEQ / 256), 256, 0, stream>>>(ctxh, denT, out, SEQ, HIDDEN, HIDDEN);
}
